// Mamba2Block_69432441307355
// MI455X (gfx1250) — matrix operand order repaired and checked by reading, not compiled, not run on hardware
//
#include <hip/hip_runtime.h>
#include <hip/hip_bf16.h>
#include <stdint.h>

// ---------------------------------------------------------------------------
// Mamba2 block for MI455X (gfx1250, wave32, WMMA).
// GEMMs: split-bf16 (3x v_wmma_f32_16x16x32_bf16), double-buffered LDS,
//        software-pipelined global->reg->LDS staging.
// Scan : async global->LDS staging of B/C rows (gfx1250 async path).
// ---------------------------------------------------------------------------

typedef __attribute__((ext_vector_type(16))) __bf16 v16bf;
typedef __attribute__((ext_vector_type(8)))  float  v8f;

#define D_MODEL   1024
#define D_STATE   64
#define N_HEADS   16
#define D_INNER   2048
#define HEAD_DIM  128
#define Z_DIM     4112      // D_INNER + 2*D_STATE*N_HEADS + N_HEADS
#define SEQ_L     4096
#define BATCH     2
#define ROWS      (BATCH * SEQ_L)   // 8192
#define RMS_EPS   1.1920928955078125e-07f

#define BM 128
#define BN 64
#define BK 32

union FragBF { v16bf v; uint32_t w[8]; unsigned short u[16]; };

__device__ __forceinline__ unsigned short f32_to_bf16_rne(float x) {
  uint32_t b = __float_as_uint(x);
  b += 0x7FFFu + ((b >> 16) & 1u);           // round-to-nearest-even
  return (unsigned short)(b >> 16);
}
__device__ __forceinline__ float bf16_bits_to_f32(unsigned short h) {
  return __uint_as_float(((uint32_t)h) << 16);
}

// Split a float4 into bf16 hi + residual-lo halves, store 8B each to LDS.
__device__ __forceinline__ void split_store(unsigned short* hi, unsigned short* lo, float4 v) {
  unsigned short h0 = f32_to_bf16_rne(v.x);
  unsigned short h1 = f32_to_bf16_rne(v.y);
  unsigned short h2 = f32_to_bf16_rne(v.z);
  unsigned short h3 = f32_to_bf16_rne(v.w);
  unsigned short l0 = f32_to_bf16_rne(v.x - bf16_bits_to_f32(h0));
  unsigned short l1 = f32_to_bf16_rne(v.y - bf16_bits_to_f32(h1));
  unsigned short l2 = f32_to_bf16_rne(v.z - bf16_bits_to_f32(h2));
  unsigned short l3 = f32_to_bf16_rne(v.w - bf16_bits_to_f32(h3));
  *(ushort4*)hi = make_ushort4(h0, h1, h2, h3);
  *(ushort4*)lo = make_ushort4(l0, l1, l2, l3);
}

// A-fragment (16x32 bf16, M x K), ISA 7.12.2 layout:
// lanes 0-15: M=lane, kb=0 ; lanes 16-31: M=lane-16, kb=8
// VGPR 0..3 : K = kb + {0,1 / 2,3 / 4,5 / 6,7}
// VGPR 4..7 : K = kb + 16 + {0,1 / 2,3 / 4,5 / 6,7}
__device__ __forceinline__ v16bf frag_a(const unsigned short* As, int lane, int mbase) {
  FragBF f;
  const unsigned short* row = As + (mbase + (lane & 15)) * BK + ((lane >> 4) << 3);
#pragma unroll
  for (int r = 0; r < 4; ++r) {
    f.w[r]     = *(const uint32_t*)(row + 2 * r);
    f.w[r + 4] = *(const uint32_t*)(row + 16 + 2 * r);
  }
  return f.v;
}

// B-fragment (32x16 bf16, K x N): N = lane&15, kb = (lane>>4)*8; elem s<8: K = kb+s, s>=8: K = kb+16+(s-8)
__device__ __forceinline__ v16bf frag_b(const unsigned short* Bs, int lane, int nbase) {
  FragBF f;
  const unsigned short* col = Bs + nbase + (lane & 15);
  const int kb = (lane >> 4) << 3;
#pragma unroll
  for (int r = 0; r < 8; ++r) {
    f.u[r]     = col[(kb + r) * BN];
    f.u[r + 8] = col[(kb + 16 + r) * BN];
  }
  return f.v;
}

struct TileRegs { float4 a[4]; float4 b[2]; };

__device__ __forceinline__ void load_tile(TileRegs& t,
    const float* __restrict__ A, const float* __restrict__ B,
    int N, int K, int m0, int n0, int k0,
    int a_r0, int a_kq, int b_r0, int b_nq)
{
#pragma unroll
  for (int it = 0; it < 4; ++it) {
    const float* src = A + (size_t)(m0 + a_r0 + (it << 5)) * K + (k0 + a_kq);
    t.a[it] = *(const float4*)src;
    if (k0 + BK < K) __builtin_prefetch(src + BK, 0, 3);
  }
#pragma unroll
  for (int it = 0; it < 2; ++it) {
    const float* src = B + (size_t)(k0 + b_r0 + (it << 4)) * N + (n0 + b_nq);
    if (n0 + b_nq + 3 < N) {
      t.b[it] = *(const float4*)src;
      if (k0 + BK < K) __builtin_prefetch(src + (size_t)BK * N, 0, 3);
    } else {
      t.b[it].x = (n0 + b_nq + 0 < N) ? src[0] : 0.f;
      t.b[it].y = (n0 + b_nq + 1 < N) ? src[1] : 0.f;
      t.b[it].z = (n0 + b_nq + 2 < N) ? src[2] : 0.f;
      t.b[it].w = (n0 + b_nq + 3 < N) ? src[3] : 0.f;
    }
  }
}

__device__ __forceinline__ void store_tile(const TileRegs& t,
    unsigned short* Ahi, unsigned short* Alo,
    unsigned short* Bhi, unsigned short* Blo,
    int a_r0, int a_kq, int b_r0, int b_nq)
{
#pragma unroll
  for (int it = 0; it < 4; ++it) {
    int m = a_r0 + (it << 5);
    split_store(&Ahi[m * BK + a_kq], &Alo[m * BK + a_kq], t.a[it]);
  }
#pragma unroll
  for (int it = 0; it < 2; ++it) {
    int kk = b_r0 + (it << 4);
    split_store(&Bhi[kk * BN + b_nq], &Blo[kk * BN + b_nq], t.b[it]);
  }
}

// C = A[M,K] @ B[K,N], row-major, M%128==0, K%32==0, N arbitrary.
// mode 0: C[idx] = acc     mode 1: Y[idx] *= sigmoid(acc)  (gate fusion)
__global__ void __launch_bounds__(256)
gemm_bf16x3(const float* __restrict__ A, const float* __restrict__ B,
            float* __restrict__ C, float* __restrict__ Y,
            int M, int N, int K, int mode)
{
  __shared__ __align__(16) unsigned short As_hi[2][BM * BK];
  __shared__ __align__(16) unsigned short As_lo[2][BM * BK];
  __shared__ __align__(16) unsigned short Bs_hi[2][BK * BN];
  __shared__ __align__(16) unsigned short Bs_lo[2][BK * BN];

  const int tid  = threadIdx.x;
  const int lane = tid & 31;
  const int wave = tid >> 5;       // 8 waves
  const int wm   = wave & 3;       // 4 waves along M (32 rows each)
  const int wn   = wave >> 2;      // 2 waves along N (32 cols each)
  const int m0   = blockIdx.y * BM;
  const int n0   = blockIdx.x * BN;

  v8f zero = {};
  v8f acc[2][2];
#pragma unroll
  for (int i = 0; i < 2; ++i)
#pragma unroll
    for (int j = 0; j < 2; ++j) acc[i][j] = zero;

  const int a_kq = (tid & 7)  * 4;   // col quad within A tile
  const int a_r0 = tid >> 3;         // 0..31
  const int b_nq = (tid & 15) * 4;   // col quad within B tile
  const int b_r0 = tid >> 4;         // 0..15

  const int nk = K / BK;

  // prologue: stage tile 0 into buffer 0
  TileRegs t0;
  load_tile(t0, A, B, N, K, m0, n0, 0, a_r0, a_kq, b_r0, b_nq);
  store_tile(t0, As_hi[0], As_lo[0], Bs_hi[0], Bs_lo[0], a_r0, a_kq, b_r0, b_nq);
  __syncthreads();

  for (int kt = 0; kt < nk; ++kt) {
    const int cur = kt & 1;
    const bool have_next = (kt + 1 < nk);

    // 1) issue global loads for tile kt+1 (latency hides under the WMMAs below)
    TileRegs tn;
    if (have_next)
      load_tile(tn, A, B, N, K, m0, n0, (kt + 1) * BK, a_r0, a_kq, b_r0, b_nq);

    // 2) compute on the current LDS buffer
    v16bf ah[2], al[2], bh[2], bl[2];
#pragma unroll
    for (int mt = 0; mt < 2; ++mt) {
      ah[mt] = frag_a(As_hi[cur], lane, wm * 32 + mt * 16);
      al[mt] = frag_a(As_lo[cur], lane, wm * 32 + mt * 16);
    }
#pragma unroll
    for (int nt = 0; nt < 2; ++nt) {
      bh[nt] = frag_b(Bs_hi[cur], lane, wn * 32 + nt * 16);
      bl[nt] = frag_b(Bs_lo[cur], lane, wn * 32 + nt * 16);
    }
#pragma unroll
    for (int mt = 0; mt < 2; ++mt)
#pragma unroll
      for (int nt = 0; nt < 2; ++nt) {
        acc[mt][nt] = __builtin_amdgcn_wmma_f32_16x16x32_bf16(
            false, ah[mt], false, bh[nt], (short)0, acc[mt][nt], false, false);
        acc[mt][nt] = __builtin_amdgcn_wmma_f32_16x16x32_bf16(
            false, ah[mt], false, bl[nt], (short)0, acc[mt][nt], false, false);
        acc[mt][nt] = __builtin_amdgcn_wmma_f32_16x16x32_bf16(
            false, al[mt], false, bh[nt], (short)0, acc[mt][nt], false, false);
      }

    // 3) convert + store tile kt+1 into the other buffer
    if (have_next) {
      const int nxt = cur ^ 1;
      store_tile(tn, As_hi[nxt], As_lo[nxt], Bs_hi[nxt], Bs_lo[nxt],
                 a_r0, a_kq, b_r0, b_nq);
    }
    __syncthreads();
  }

  // ---- store (C/D layout: lane<16 -> M=v, lane>=16 -> M=v+8; N = lane&15) ----
  const int cm = (lane >> 4) << 3;
  const int cn = lane & 15;
#pragma unroll
  for (int mt = 0; mt < 2; ++mt)
#pragma unroll
    for (int nt = 0; nt < 2; ++nt) {
      int col = n0 + wn * 32 + nt * 16 + cn;
      if (col < N) {
        int rowb = m0 + wm * 32 + mt * 16 + cm;
#pragma unroll
        for (int v = 0; v < 8; ++v) {
          size_t idx = (size_t)(rowb + v) * N + col;
          float val = acc[mt][nt][v];
          if (mode == 1) {
            Y[idx] *= 1.f / (1.f + __expf(-val));   // gate fusion
          } else {
            C[idx] = val;
          }
        }
      }
    }
}

// ---- causal depthwise conv (4 taps) + SiLU:  z[:, :, :D_INNER] -> xs ----
__global__ void __launch_bounds__(256)
conv_silu_kernel(const float* __restrict__ z, const float* __restrict__ conv_w,
                 const float* __restrict__ conv_b, float* __restrict__ xs)
{
  int idx = blockIdx.x * 256 + threadIdx.x;     // 8192*2048 total
  int c  = idx & (D_INNER - 1);
  int bl = idx >> 11;                           // b*L + l
  int l  = bl & (SEQ_L - 1);
  float acc = conv_b[c];
#pragma unroll
  for (int k = 0; k < 4; ++k) {
    int ls = l - 3 + k;
    if (ls >= 0) acc += z[(size_t)(bl - 3 + k) * Z_DIM + c] * conv_w[c * 4 + k];
  }
  xs[idx] = acc / (1.f + __expf(-acc));         // SiLU
}

// ---- sequential SSM scan: one WG per (batch, head); state in registers ----
// thread t: d = t>>1 (0..127), owns s-range [(t&1)*32, +32) of the 128x64 state.
// B_t / C_t rows staged via gfx1250 async global->LDS copies (no VGPR round
// trip on the serial critical path), published with s_wait_asynccnt + barrier.
__global__ void __launch_bounds__(256)
scan_kernel(const float* __restrict__ z, const float* __restrict__ xs,
            const float* __restrict__ delta_param, float* __restrict__ y)
{
  __shared__ __align__(16) float bsh[D_STATE];
  __shared__ __align__(16) float csh[D_STATE];
  const int b    = blockIdx.x >> 4;
  const int h    = blockIdx.x & 15;
  const int tid  = threadIdx.x;
  const int lane = tid & 31;
  const int d    = tid >> 1;
  const int sh   = (tid & 1) << 5;

  float hs[32];
#pragma unroll
  for (int i = 0; i < 32; ++i) hs[i] = 0.f;

  const float dpar = delta_param[h];
  const float* zb = z  + (size_t)b * SEQ_L * Z_DIM;
  const float* xb = xs + (size_t)b * SEQ_L * D_INNER + h * HEAD_DIM;
  float*       yb = y  + (size_t)b * SEQ_L * D_INNER + h * HEAD_DIM;

  // lane-parallel async copy setup (wave 0 only):
  // lanes 0-15: 16B each of B_t -> bsh ; lanes 16-31: 16B each of C_t -> csh
  const uint32_t lds_dst = (lane < 16)
      ? ((uint32_t)(uintptr_t)bsh + (uint32_t)lane * 16u)
      : ((uint32_t)(uintptr_t)csh + (uint32_t)(lane - 16) * 16u);
  const size_t g_off = (lane < 16)
      ? (size_t)(D_INNER + h * D_STATE + lane * 4)
      : (size_t)(D_INNER + N_HEADS * D_STATE + h * D_STATE + (lane - 16) * 4);

  for (int l = 0; l < SEQ_L; ++l) {
    const float* zrow = zb + (size_t)l * Z_DIM;
    if (tid < 32) {
      uint64_t gaddr = (uint64_t)(uintptr_t)(zrow + g_off);
      asm volatile("global_load_async_to_lds_b128 %0, %1, off"
                   :: "v"(lds_dst), "v"(gaddr) : "memory");
      asm volatile("s_wait_asynccnt 0x0" ::: "memory");
    }
    float dtv   = zrow[D_INNER + 2 * N_HEADS * D_STATE + h];
    float spa   = dtv + dpar;
    float delta = (spa > 20.f) ? spa : log1pf(__expf(spa));  // softplus
    float a     = __expf(-delta * fabsf(dtv));
    float xd    = xb[(size_t)l * D_INNER + d];
    __syncthreads();

    const float4* bp = (const float4*)(bsh + sh);
    const float4* cp = (const float4*)(csh + sh);
    float accp = 0.f;
#pragma unroll
    for (int i = 0; i < 8; ++i) {
      float4 bv = bp[i], cv = cp[i];
      hs[4*i+0] = fmaf(a, hs[4*i+0], xd * bv.x); accp = fmaf(hs[4*i+0], cv.x, accp);
      hs[4*i+1] = fmaf(a, hs[4*i+1], xd * bv.y); accp = fmaf(hs[4*i+1], cv.y, accp);
      hs[4*i+2] = fmaf(a, hs[4*i+2], xd * bv.z); accp = fmaf(hs[4*i+2], cv.z, accp);
      hs[4*i+3] = fmaf(a, hs[4*i+3], xd * bv.w); accp = fmaf(hs[4*i+3], cv.w, accp);
    }
    float tot = accp + __shfl_xor(accp, 1, 32);   // combine the two s-halves
    if ((tid & 1) == 0) yb[(size_t)l * D_INNER + d] = tot;
    __syncthreads();
  }
}

// ---- RMS norm over last dim (2048), per-row workgroup ----
__global__ void __launch_bounds__(256)
rmsnorm_kernel(const float* __restrict__ y, const float* __restrict__ norm_w,
               float* __restrict__ t)
{
  const int row = blockIdx.x;
  const int tid = threadIdx.x;
  const float* yr = y + (size_t)row * D_INNER;
  float v[8];
  float ss = 0.f;
#pragma unroll
  for (int i = 0; i < 8; ++i) { v[i] = yr[tid + (i << 8)]; ss = fmaf(v[i], v[i], ss); }
#pragma unroll
  for (int off = 16; off > 0; off >>= 1) ss += __shfl_xor(ss, off, 32);
  __shared__ float wred[8];
  if ((tid & 31) == 0) wred[tid >> 5] = ss;
  __syncthreads();
  float tot = 0.f;
#pragma unroll
  for (int w = 0; w < 8; ++w) tot += wred[w];
  float scale = rsqrtf(tot * (1.f / D_INNER) + RMS_EPS);
  float* tr = t + (size_t)row * D_INNER;
#pragma unroll
  for (int i = 0; i < 8; ++i) { int c = tid + (i << 8); tr[c] = v[i] * scale * norm_w[c]; }
}

// ---------------------------------------------------------------------------
extern "C" void kernel_launch(void* const* d_in, const int* in_sizes, int n_in,
                              void* d_out, int out_size, void* d_ws, size_t ws_size,
                              hipStream_t stream) {
  const float* u           = (const float*)d_in[0];
  const float* W_in        = (const float*)d_in[1];
  const float* conv_w      = (const float*)d_in[2];
  const float* conv_b      = (const float*)d_in[3];
  const float* delta_param = (const float*)d_in[4];
  const float* W_gate      = (const float*)d_in[5];
  const float* norm_w      = (const float*)d_in[6];
  const float* W_out       = (const float*)d_in[7];
  float* out = (float*)d_out;

  float* z  = (float*)d_ws;                         // 8192 x 4112
  float* xs = z  + (size_t)ROWS * Z_DIM;            // 8192 x 2048 (conv out, later normed y)
  float* y  = xs + (size_t)ROWS * D_INNER;          // 8192 x 2048 (scan out, then gated)

  dim3 blk(256);

  // 1. z = u @ W_in                      (8192 x 1024 x 4112)
  gemm_bf16x3<<<dim3((Z_DIM + BN - 1) / BN, ROWS / BM), blk, 0, stream>>>(
      u, W_in, z, nullptr, ROWS, Z_DIM, D_MODEL, 0);

  // 2. xs = silu(conv(z[..., :2048]))
  conv_silu_kernel<<<(ROWS * D_INNER) / 256, blk, 0, stream>>>(z, conv_w, conv_b, xs);

  // 3. y = ssm_scan(xs, z[B,C,dt])       (32 WGs, serial in L)
  scan_kernel<<<BATCH * N_HEADS, blk, 0, stream>>>(z, xs, delta_param, y);

  // 4. y *= sigmoid(u @ W_gate)          (8192 x 1024 x 2048, fused epilogue)
  gemm_bf16x3<<<dim3(D_INNER / BN, ROWS / BM), blk, 0, stream>>>(
      u, W_gate, nullptr, y, ROWS, D_INNER, D_MODEL, 1);

  // 5. xs = rmsnorm(y) * norm_w          (reuse xs buffer)
  rmsnorm_kernel<<<ROWS, blk, 0, stream>>>(y, norm_w, xs);

  // 6. out = xs @ W_out                  (8192 x 2048 x 1024)
  gemm_bf16x3<<<dim3(D_MODEL / BN, ROWS / BM), blk, 0, stream>>>(
      xs, W_out, out, nullptr, ROWS, D_MODEL, D_INNER, 0);
}
